// Attention_light_dwconv_v3_50714973831583
// MI455X (gfx1250) — compile-verified, matrix operand order checked by reading, not run on hardware
//
#include <hip/hip_runtime.h>
#include <hip/hip_bf16.h>
#include <math.h>

typedef _Float16 f16;
typedef __attribute__((ext_vector_type(16))) _Float16 v16h;
typedef __attribute__((ext_vector_type(8)))  float    v8f;
typedef __attribute__((ext_vector_type(8)))  unsigned int v8u;
typedef __attribute__((ext_vector_type(4)))  unsigned int u32x4;
typedef __attribute__((ext_vector_type(8)))  int          i32x8;
typedef __attribute__((ext_vector_type(4)))  int          i32x4;

#define B_SZ   32
#define N_Q    3136
#define DIM_C  320
#define HEADS  5
#define CNEW   400
#define NKV    196
#define DQH    80      // q/k head dim
#define DQP    96      // q head-dim padded (3 k-steps of 32)
#define KLD    104     // k row stride (f16): 52 dwords -> 16 distinct LDS banks
#define DVH    64
#define NKP    208     // padded key rows (13*16)
#define VLD    232     // v row stride (f16): 116 dwords -> 16 distinct LDS banks
#define PLD    232     // P LDS stride
#define QSCALE 0.11180339887498948f   // 80^-0.5

// ---- fragment helpers ------------------------------------------------------
__device__ __forceinline__ v16h as_h16(v8u u) {
  union { v8u u8; v16h h16; } c; c.u8 = u; return c.h16;
}

// Unguarded 16x32 f16 A-style fragment (also B as A-of-B^T) from a row-major
// f16 matrix (global or LDS). ISA layout: lane m=lane&15, hh=lane>>4; VGPR v
// holds the f16 pair at K = k0 + (v&4?16:0) + 8*hh + 2*(v&3). VGPR0-3 and
// VGPR4-7 are each 16 contiguous bytes -> two 128-bit loads. Caller ensures:
//   ld % 8 == 0, k0 % 32 == 0, base 16B-aligned, all accesses in-bounds.
__device__ __forceinline__ v8u ldfrag_u(const f16* __restrict__ p, int ld,
                                        int row0, int k0) {
  const int lane = threadIdx.x & 31;
  const int m  = row0 + (lane & 15);
  const int hh = lane >> 4;
  const f16* rp = p + (size_t)m * ld + k0 + hh * 8;
  const uint4 lo = *(const uint4*)(rp);
  const uint4 hi = *(const uint4*)(rp + 16);
  v8u r;
  r[0] = lo.x; r[1] = lo.y; r[2] = lo.z; r[3] = lo.w;
  r[4] = hi.x; r[5] = hi.y; r[6] = hi.z; r[7] = hi.w;
  return r;
}

__device__ __forceinline__ v8f wmma32(v8u a, v8u b, v8f c) {
  return __builtin_amdgcn_wmma_f32_16x16x32_f16(
      false, as_h16(a), false, as_h16(b), (short)0, c, false, false);
}

// Issue a 1-D TDM copy of nelem f16 from global -> LDS (contiguous).
// D# layout per CDNA5 ISA ch.8: group0 = {count=1 | lds_addr | gaddr | type=2},
// group1 = {data_size=2B | tensor_dim0=tile_dim0=nelem | dim0_stride=nelem}.
// This toolchain exposes the 6-arg builtin: (g0, g1, g2, g3, g4, cpol);
// groups 2..4 are zero (<=2-D tile).
__device__ __forceinline__ void tdm_load_1d(unsigned lds_off, const void* gptr,
                                            unsigned nelem) {
  const unsigned long long ga = (unsigned long long)gptr;
  u32x4 g0;
  g0[0] = 1u;                                              // count=1 (user D#)
  g0[1] = lds_off;                                         // lds_addr (bytes)
  g0[2] = (unsigned)ga;                                    // global_addr[31:0]
  g0[3] = ((unsigned)(ga >> 32) & 0x01FFFFFFu) | (2u << 30); // [56:32] | type=2
  i32x8 g1;
  g1[0] = (int)(1u << 16);                                 // data_size=1 (2B)
  g1[1] = (int)((nelem & 0xFFFFu) << 16);                  // tensor_dim0[15:0]
  g1[2] = (int)((nelem >> 16) | (1u << 16));               // dim0[31:16], dim1=1
  g1[3] = (int)((nelem & 0xFFFFu) << 16);                  // tile_dim0
  g1[4] = 0;                                               // tile_dim1/2 unused
  g1[5] = (int)nelem;                                      // dim0_stride[31:0]
  g1[6] = 0;
  g1[7] = 0;
  const i32x4 z4 = {0, 0, 0, 0};
  const i32x8 z8 = {0, 0, 0, 0, 0, 0, 0, 0};
  __builtin_amdgcn_tensor_load_to_lds(g0, g1, z4, z4, z8, 0);
}

// ---- generic WMMA GEMM: C[M,N] = A[M,Kp](f16) * Wt[Np,Kp]^T(f16) -----------
// 256 threads = 8 waves (4x2). Block tile 128x64, wave tile 32x32 (2x2 WMMAs).
// M multiple of 128; A/Wt zero-padded to Kp cols / gridDim.x*64 rows.
// Epilogue MODEs:
//   0: f32 out [M,N] + bias       1: f16 out [M,N]
//   2: q head-scatter -> qh[B,H,3136,96]
//   3: v transpose-scatter -> vT[B,H,64,232]
//   4: k head-scatter -> kh[B,H,208,104]
template <int MODE>
__global__ __launch_bounds__(256) void gemm_wmma(
    const f16* __restrict__ A, const f16* __restrict__ Wt,
    const float* __restrict__ bias, float* __restrict__ Cf,
    f16* __restrict__ Ch, int M, int N, int Kp, float scale)
{
  const int lane = threadIdx.x & 31;
  const int wave = threadIdx.x >> 5;
  const int rowBase = blockIdx.y * 128 + (wave >> 1) * 32;
  const int colBase = blockIdx.x * 64  + (wave & 1)  * 32;

  v8f acc[2][2];
#pragma unroll
  for (int i = 0; i < 2; ++i)
#pragma unroll
    for (int j = 0; j < 2; ++j)
#pragma unroll
      for (int e = 0; e < 8; ++e) acc[i][j][e] = 0.f;

  for (int k0 = 0; k0 < Kp; k0 += 32) {
    v8u a0 = ldfrag_u(A  + (size_t)rowBase * Kp,        Kp, 0,  k0);
    v8u a1 = ldfrag_u(A  + (size_t)(rowBase + 16) * Kp, Kp, 0,  k0);
    v8u b0 = ldfrag_u(Wt + (size_t)colBase * Kp,        Kp, 0,  k0);
    v8u b1 = ldfrag_u(Wt + (size_t)(colBase + 16) * Kp, Kp, 0,  k0);
    acc[0][0] = wmma32(a0, b0, acc[0][0]);
    acc[0][1] = wmma32(a0, b1, acc[0][1]);
    acc[1][0] = wmma32(a1, b0, acc[1][0]);
    acc[1][1] = wmma32(a1, b1, acc[1][1]);
  }

  const int hh = lane >> 4, ln = lane & 15;
#pragma unroll
  for (int i = 0; i < 2; ++i)
#pragma unroll
    for (int j = 0; j < 2; ++j)
#pragma unroll
      for (int e = 0; e < 8; ++e) {
        const int r = rowBase + i * 16 + e + 8 * hh;   // always < M
        const int c = colBase + j * 16 + ln;
        if (c < N) {
          float val = acc[i][j][e] * scale;
          if (MODE == 0) {
            Cf[(size_t)r * N + c] = val + bias[c];
          } else if (MODE == 1) {
            Ch[(size_t)r * N + c] = (f16)val;
          } else if (MODE == 2) {
            const int b = r / N_Q, rr = r % N_Q;
            const int h = c / DQH, kk = c % DQH;
            Ch[((size_t)(b * HEADS + h) * N_Q + rr) * DQP + kk] = (f16)val;
          } else if (MODE == 3) {
            const int b = r / NKV, kk = r % NKV;
            const int h = c / DVH, n = c % DVH;
            Ch[((size_t)(b * HEADS + h) * DVH + n) * VLD + kk] = (f16)val;
          } else { // MODE == 4
            const int b = r / NKV, rr = r % NKV;
            const int h = c / DQH, kk = c % DQH;
            Ch[((size_t)(b * HEADS + h) * NKP + rr) * KLD + kk] = (f16)val;
          }
        }
      }
}

// ---- fused attention: softmax(q k^T) v per (b, h), 64 q-rows per block -----
// TDM stages the whole K tile (208x104 f16) and V tile (64x232 f16) for this
// (b,h) into LDS once per block (wave 0 issues, TENSORcnt-waited, barrier).
// 4 waves x 16 q-rows; 16x208 score strip in 13 WMMA f32 accumulators;
// softmax via width-16 shuffles; P staged through LDS (stride 232) to turn
// the C layout into A fragments for the P*V WMMA chain.
__global__ __launch_bounds__(128) void attn_kernel(
    const f16* __restrict__ qh, const f16* __restrict__ kh,
    const f16* __restrict__ vT, f16* __restrict__ o)
{
  __shared__ __align__(16) f16 Kt[NKP * KLD];    // 43,264 B
  __shared__ __align__(16) f16 Vt[DVH * VLD];    // 29,696 B
  __shared__ __align__(16) f16 Plds[64 * PLD];   // 29,696 B
  const int lane = threadIdx.x & 31;
  const int wave = threadIdx.x >> 5;
  const int b = blockIdx.z, h = blockIdx.y;
  const int qrow = blockIdx.x * 64 + wave * 16;   // row within (b, h)
  const int hh = lane >> 4, ln = lane & 15;

  const f16* qbase = qh + (size_t)(b * HEADS + h) * N_Q * DQP;

  // ---- TDM: K and V tiles -> LDS (one DMA pair per block, wave 0 only)
  const int wid = __builtin_amdgcn_readfirstlane((int)threadIdx.x) >> 5;
  if (wid == 0) {
    tdm_load_1d((unsigned)(size_t)&Kt[0],
                kh + (size_t)(b * HEADS + h) * NKP * KLD, NKP * KLD);
    tdm_load_1d((unsigned)(size_t)&Vt[0],
                vT + (size_t)(b * HEADS + h) * DVH * VLD, DVH * VLD);
    __builtin_amdgcn_s_wait_tensorcnt(0);
  }
  __syncthreads();

  // ---- stage 1: S = (q*scale) K^T; 13 n-tiles, K-dim 80 zero-padded to 96
  v8f s[13];
#pragma unroll
  for (int t = 0; t < 13; ++t)
#pragma unroll
    for (int e = 0; e < 8; ++e) s[t][e] = 0.f;

#pragma unroll
  for (int ks = 0; ks < 3; ++ks) {
    const int k0 = ks * 32;
    const v8u a = ldfrag_u(qbase, DQP, qrow, k0);
#pragma unroll
    for (int t = 0; t < 13; ++t) {
      const v8u bf = ldfrag_u(Kt, KLD, t * 16, k0);
      s[t] = wmma32(a, bf, s[t]);
    }
  }

  // ---- stage 2: softmax over n (196 valid of 208), fully in registers
  const bool nval12 = (ln < 4);   // tile 12: n = 192+ln valid iff < 196
#pragma unroll
  for (int e = 0; e < 8; ++e) {
    float mx = -1e30f;
#pragma unroll
    for (int t = 0; t < 13; ++t) {
      float v = s[t][e];
      if (t == 12 && !nval12) v = -1e30f;
      mx = fmaxf(mx, v);
    }
    mx = fmaxf(mx, __shfl_xor(mx, 8, 16));
    mx = fmaxf(mx, __shfl_xor(mx, 4, 16));
    mx = fmaxf(mx, __shfl_xor(mx, 2, 16));
    mx = fmaxf(mx, __shfl_xor(mx, 1, 16));
    float sum = 0.f;
#pragma unroll
    for (int t = 0; t < 13; ++t) {
      float v = (t == 12 && !nval12) ? 0.f : __expf(s[t][e] - mx);
      s[t][e] = v;
      sum += v;
    }
    sum += __shfl_xor(sum, 8, 16);
    sum += __shfl_xor(sum, 4, 16);
    sum += __shfl_xor(sum, 2, 16);
    sum += __shfl_xor(sum, 1, 16);
    const float rs = 1.f / sum;
#pragma unroll
    for (int t = 0; t < 13; ++t) s[t][e] *= rs;
  }

  // ---- spill P (f16) to LDS; zero cols 208..231 for this wave's rows
#pragma unroll
  for (int e = 0; e < 8; ++e) {
    const int m = wave * 16 + e + 8 * hh;
#pragma unroll
    for (int t = 0; t < 13; ++t)
      Plds[m * PLD + t * 16 + ln] = (f16)s[t][e];
  }
  for (int i = lane; i < 16 * 24; i += 32) {
    const int rr = wave * 16 + i / 24, cc = 208 + i % 24;
    Plds[rr * PLD + cc] = (f16)0.f;
  }
  __syncthreads();

  // ---- stage 3: O = P V; K-dim 196 zero-padded to 224 (7 k-steps), N = 64
  v8f oacc[4];
#pragma unroll
  for (int j = 0; j < 4; ++j)
#pragma unroll
    for (int e = 0; e < 8; ++e) oacc[j][e] = 0.f;

#pragma unroll
  for (int ks = 0; ks < 7; ++ks) {
    const int k0 = ks * 32;
    const v8u a = ldfrag_u(Plds, PLD, wave * 16, k0);
#pragma unroll
    for (int j = 0; j < 4; ++j) {
      const v8u bf = ldfrag_u(Vt, VLD, j * 16, k0);
      oacc[j] = wmma32(a, bf, oacc[j]);
    }
  }

#pragma unroll
  for (int j = 0; j < 4; ++j)
#pragma unroll
    for (int e = 0; e < 8; ++e) {
      const int r = b * N_Q + qrow + e + 8 * hh;
      const int c = h * DVH + j * 16 + ln;
      o[(size_t)r * DIM_C + c] = (f16)oacc[j][e];
    }
}

// ---- small prep / pointwise kernels ----------------------------------------
__global__ void to_f16_kernel(const float* __restrict__ s, f16* __restrict__ d, long n) {
  long i = (long)blockIdx.x * 256 + threadIdx.x;
  if (i < n) d[i] = (f16)s[i];
}

__global__ void zero_f16_kernel(f16* __restrict__ p, long n) {
  long i = (long)blockIdx.x * 256 + threadIdx.x;
  if (i < n) p[i] = (f16)0.f;
}

// zero columns [c0, ld) of a [rows, ld] f16 matrix
__global__ void zero_tail_kernel(f16* __restrict__ p, long rows, int ld, int c0) {
  const int span = ld - c0;
  const long n = rows * span;
  long i = (long)blockIdx.x * 256 + threadIdx.x;
  if (i < n) {
    const long row = i / span;
    p[row * ld + c0 + (int)(i % span)] = (f16)0.f;
  }
}

// src [K,N] f32 -> dst [Np,Kp] f16 (transposed, zero-padded)
__global__ void transpose_pad_f16(const float* __restrict__ s, f16* __restrict__ d,
                                  int K, int N, int Kp, int Np) {
  const int idx = blockIdx.x * 256 + threadIdx.x;
  if (idx >= Np * Kp) return;
  const int n = idx / Kp, k = idx % Kp;
  float v = 0.f;
  if (n < N && k < K) v = s[(size_t)k * N + n];
  d[idx] = (f16)v;
}

// src [N,K] f32 -> dst [Np,Kp] f16 (already [N,K]; just pad + convert)
__global__ void convert_pad_f16(const float* __restrict__ s, f16* __restrict__ d,
                                int K, int N, int Kp, int Np) {
  const int idx = blockIdx.x * 256 + threadIdx.x;
  if (idx >= Np * Kp) return;
  const int n = idx / Kp, k = idx % Kp;
  float v = 0.f;
  if (n < N && k < K) v = s[(size_t)n * K + k];
  d[idx] = (f16)v;
}

// depthwise 4x4 stride-4 conv: x [B,3136,320] -> t [B,196,320] f16
__global__ void dwconv_kernel(const float* __restrict__ x, const float* __restrict__ w,
                              const float* __restrict__ bias, f16* __restrict__ t) {
  const int idx = blockIdx.x * 256 + threadIdx.x;
  if (idx >= B_SZ * NKV * DIM_C) return;
  const int c = idx % DIM_C;
  const int p = (idx / DIM_C) % NKV;
  const int b = idx / (DIM_C * NKV);
  const int oy = p / 14, ox = p % 14;
  float acc = bias[c];
  const float* xb = x + (size_t)b * N_Q * DIM_C;
#pragma unroll
  for (int ky = 0; ky < 4; ++ky)
#pragma unroll
    for (int kx = 0; kx < 4; ++kx) {
      const int n = (oy * 4 + ky) * 56 + ox * 4 + kx;
      acc += xb[(size_t)n * DIM_C + c] * w[c * 16 + ky * 4 + kx];
    }
  t[idx] = (f16)acc;
}

// LayerNorm(400) + exact GELU; writes f16 rows with ld=416, pad zeroed
__global__ __launch_bounds__(128) void ln_gelu_kernel(
    const float* __restrict__ xin, const float* __restrict__ g,
    const float* __restrict__ be, f16* __restrict__ out) {
  __shared__ float red[128];
  const int row = blockIdx.x, tid = threadIdx.x;
  const float* xr = xin + (size_t)row * CNEW;
  float s = 0.f;
  for (int c = tid; c < CNEW; c += 128) s += xr[c];
  red[tid] = s; __syncthreads();
  for (int st = 64; st > 0; st >>= 1) { if (tid < st) red[tid] += red[tid + st]; __syncthreads(); }
  const float mu = red[0] / CNEW;
  __syncthreads();
  float vs = 0.f;
  for (int c = tid; c < CNEW; c += 128) { float d = xr[c] - mu; vs += d * d; }
  red[tid] = vs; __syncthreads();
  for (int st = 64; st > 0; st >>= 1) { if (tid < st) red[tid] += red[tid + st]; __syncthreads(); }
  const float inv = rsqrtf(red[0] / CNEW + 1e-5f);
  for (int c = tid; c < CNEW; c += 128) {
    const float y = (xr[c] - mu) * inv * g[c] + be[c];
    const float ge = 0.5f * y * (1.f + erff(y * 0.70710678118f));
    out[(size_t)row * 416 + c] = (f16)ge;
  }
  if (tid < 16) out[(size_t)row * 416 + CNEW + tid] = (f16)0.f;  // K-pad
}

// ---- launch ----------------------------------------------------------------
extern "C" void kernel_launch(void* const* d_in, const int* in_sizes, int n_in,
                              void* d_out, int out_size, void* d_ws, size_t ws_size,
                              hipStream_t stream) {
  const float* x      = (const float*)d_in[0];
  const float* dw_w   = (const float*)d_in[3];
  const float* dw_b   = (const float*)d_in[4];
  const float* pw_w   = (const float*)d_in[5];   // [400,320] already [N,K]
  const float* pw_b   = (const float*)d_in[6];
  const float* ln_g   = (const float*)d_in[7];
  const float* ln_b   = (const float*)d_in[8];
  const float* q_w    = (const float*)d_in[9];   // [320,400] -> transpose
  const float* k_w    = (const float*)d_in[10];  // [400,400] -> transpose
  const float* v_w    = (const float*)d_in[11];  // [400,320] -> transpose
  const float* proj_w = (const float*)d_in[12];  // [320,320] -> transpose
  const float* proj_b = (const float*)d_in[13];
  float* out = (float*)d_out;

  char* base = (char*)d_ws;
  size_t off = 0;
  auto alloc = [&](size_t bytes) {
    void* p = base + off;
    off = (off + bytes + 255) & ~(size_t)255;
    return p;
  };

  const long QH_ROWS = (long)B_SZ * HEADS * N_Q;         // 501,760
  const long KH_ELTS = (long)B_SZ * HEADS * NKP * KLD;   // 3,461,120
  const long VT_ELTS = (long)B_SZ * HEADS * DVH * VLD;   // 2,375,680

  f16*   xh     = (f16*)alloc((size_t)100352 * 320 * 2);  // x f16; reused as attn_out
  f16*   qhb    = (f16*)alloc((size_t)QH_ROWS * DQP * 2); // q head-major padded
  f16*   tbuf   = (f16*)alloc((size_t)6272 * 320 * 2);
  float* xs_pre = (float*)alloc((size_t)6272 * 400 * 4);
  f16*   xs     = (f16*)alloc((size_t)6272 * 416 * 2);    // K padded to 416
  f16*   khb    = (f16*)alloc((size_t)KH_ELTS * 2);       // k head-major padded
  f16*   vTb    = (f16*)alloc((size_t)VT_ELTS * 2);       // v transposed padded
  f16*   wq_t   = (f16*)alloc((size_t)448 * 320 * 2);
  f16*   wk_t   = (f16*)alloc((size_t)448 * 416 * 2);
  f16*   wv_t   = (f16*)alloc((size_t)320 * 416 * 2);
  f16*   wp_t   = (f16*)alloc((size_t)320 * 320 * 2);
  f16*   pw_t   = (f16*)alloc((size_t)448 * 320 * 2);

  // precision / layout prep
  to_f16_kernel<<<(32112640 + 255) / 256, 256, 0, stream>>>(x, xh, 32112640L);
  transpose_pad_f16<<<(448 * 320 + 255) / 256, 256, 0, stream>>>(q_w,    wq_t, 320, 400, 320, 448);
  transpose_pad_f16<<<(448 * 416 + 255) / 256, 256, 0, stream>>>(k_w,    wk_t, 400, 400, 416, 448);
  transpose_pad_f16<<<(320 * 416 + 255) / 256, 256, 0, stream>>>(v_w,    wv_t, 400, 320, 416, 320);
  transpose_pad_f16<<<(320 * 320 + 255) / 256, 256, 0, stream>>>(proj_w, wp_t, 320, 320, 320, 320);
  convert_pad_f16<<<(448 * 320 + 255) / 256, 256, 0, stream>>>(pw_w, pw_t, 320, 400, 320, 448);

  // zero-init padded scatter targets (scatters never touch pad regions)
  zero_f16_kernel<<<(int)((KH_ELTS + 255) / 256), 256, 0, stream>>>(khb, KH_ELTS);
  zero_f16_kernel<<<(int)((VT_ELTS + 255) / 256), 256, 0, stream>>>(vTb, VT_ELTS);
  zero_tail_kernel<<<(int)((QH_ROWS * 16 + 255) / 256), 256, 0, stream>>>(qhb, QH_ROWS, DQP, DQH);

  // spatial reduction path
  dwconv_kernel<<<(B_SZ * NKV * DIM_C + 255) / 256, 256, 0, stream>>>(x, dw_w, dw_b, tbuf);
  gemm_wmma<0><<<dim3(7, 49), 256, 0, stream>>>(
      tbuf, pw_t, pw_b, xs_pre, nullptr, 6272, 400, 320, 1.f);
  ln_gelu_kernel<<<6272, 128, 0, stream>>>(xs_pre, ln_g, ln_b, xs);

  // projections (epilogues scatter straight into padded attention layouts)
  gemm_wmma<4><<<dim3(7, 49), 256, 0, stream>>>(
      xs, wk_t, nullptr, nullptr, khb, 6272, 400, 416, 1.f);
  gemm_wmma<3><<<dim3(5, 49), 256, 0, stream>>>(
      xs, wv_t, nullptr, nullptr, vTb, 6272, 320, 416, 1.f);
  gemm_wmma<2><<<dim3(7, 784), 256, 0, stream>>>(
      xh, wq_t, nullptr, nullptr, qhb, 100352, 400, 320, QSCALE);

  // fused attention (TDM-staged K/V; writes attn_out into xh)
  attn_kernel<<<dim3(49, HEADS, B_SZ), 128, 0, stream>>>(qhb, khb, vTb, xh);

  // output projection
  gemm_wmma<0><<<dim3(5, 784), 256, 0, stream>>>(
      xh, wp_t, proj_b, out, nullptr, 100352, 320, 320, 1.f);
}
